// GLAMemoryCompressor_7456063226407
// MI455X (gfx1250) — matrix operand order repaired and checked by reading, not compiled, not run on hardware
//
#include <hip/hip_runtime.h>
#include <hip/hip_bf16.h>
#include <math.h>

// ---------------- problem constants (from reference) ----------------
#define B_    4
#define T_    2048
#define D_    2048
#define KDIM  1024
#define VDIM  2048
#define R_    16
#define H_    8
#define DK_   128
#define DV_   256
#define CHUNK_ 64
#define NC_   32
#define BT_   (B_ * T_)      // 8192
#define GLN_  16.0f
#define EPS_  1e-5f

typedef __attribute__((ext_vector_type(16))) __bf16 v16bf;
typedef __attribute__((ext_vector_type(8)))  float  v8f;

union V16Frag {
    uint4  u[2];
    v16bf  v;
};

// ---------------------------------------------------------------
// Tensor Data Mover support (probe-verified builtin, guarded).
// ---------------------------------------------------------------
#if __has_builtin(__builtin_amdgcn_tensor_load_to_lds) && __has_builtin(__builtin_amdgcn_s_wait_tensorcnt)
#define HAVE_TDM 1
#else
#define HAVE_TDM 0
#endif

#if HAVE_TDM
typedef unsigned int u32x4 __attribute__((ext_vector_type(4)));
typedef int          i32x4 __attribute__((ext_vector_type(4)));
typedef int          i32x8 __attribute__((ext_vector_type(8)));

// Flat LDS aperture: low 32 bits of a generic pointer to LDS are the LDS byte
// offset (ISA 10.2 aperture mapping).
__device__ __forceinline__ unsigned lds_off_of(void* p)
{
    return (unsigned)(unsigned long long)(uintptr_t)p;
}

// 2-D TDM load: tile_rows x tile_cols elements (row-major, packed into LDS).
// elem_code: 0=1B 1=2B 2=4B 3=8B.  stride/tensor dims in elements.
__device__ __forceinline__ void tdm_load_2d(const void* gsrc, void* ldst,
                                            unsigned tile_cols, unsigned tile_rows,
                                            unsigned tensor_cols, unsigned tensor_rows,
                                            unsigned stride_elems, unsigned elem_code)
{
    unsigned long long ga = (unsigned long long)(uintptr_t)gsrc;
    u32x4 g0;
    g0[0] = 1u;                                               // count=1, user D#
    g0[1] = lds_off_of(ldst);                                 // lds_addr [63:32]
    g0[2] = (unsigned)(ga & 0xFFFFFFFFu);                     // global_addr lo
    g0[3] = (unsigned)((ga >> 32) & 0x01FFFFFFu) | (2u << 30);// addr hi | type=2
    i32x8 g1;
    g1[0] = (int)(elem_code << 16);                           // wg_mask=0, data_size
    g1[1] = (int)((tensor_cols & 0xFFFFu) << 16);             // abar=0 | tdim0 lo16
    g1[2] = (int)((tensor_cols >> 16) | ((tensor_rows & 0xFFFFu) << 16));
    g1[3] = (int)((tensor_rows >> 16) | (tile_cols << 16));   // tdim1 hi | tile_dim0
    g1[4] = (int)(tile_rows & 0xFFFFu);                       // tile_dim1 | tile_dim2=0
    g1[5] = (int)stride_elems;                                // dim0_stride lo32
    g1[6] = 0;                                                // stride hi | dim1_stride
    g1[7] = 0;
    i32x4 z4 = {0, 0, 0, 0};
#if __clang_major__ >= 23
    i32x8 z8 = {0, 0, 0, 0, 0, 0, 0, 0};
    __builtin_amdgcn_tensor_load_to_lds(g0, g1, z4, z4, z8, 0);
#else
    __builtin_amdgcn_tensor_load_to_lds(g0, g1, z4, z4, 0);
#endif
}
#endif // HAVE_TDM

// ---------------------------------------------------------------
// Kernel 1: RMSNorm (f32) -> x in bf16.  One block per row of 2048.
// ---------------------------------------------------------------
__global__ __launch_bounds__(256) void k_rmsnorm_bf16(const float* __restrict__ seg,
                                                      const float* __restrict__ nw,
                                                      __bf16* __restrict__ xout)
{
    const int row = blockIdx.x;                 // 0..8191
    const float* src = seg + (size_t)row * D_;
    __shared__ float red[256];

    float ss = 0.f;
#pragma unroll
    for (int j = 0; j < 8; ++j) {
        float v = src[threadIdx.x + j * 256];
        ss += v * v;
    }
    red[threadIdx.x] = ss;
    __syncthreads();
    for (int s = 128; s > 0; s >>= 1) {
        if (threadIdx.x < s) red[threadIdx.x] += red[threadIdx.x + s];
        __syncthreads();
    }
    const float scale = rsqrtf(red[0] * (1.0f / (float)D_) + EPS_);

#pragma unroll
    for (int j = 0; j < 8; ++j) {
        int c = threadIdx.x + j * 256;
        xout[(size_t)row * D_ + c] = (__bf16)(src[c] * scale * nw[c]);
    }
}

// ---------------------------------------------------------------
// Kernel 2: f32 -> bf16 convert (grid-stride).
// ---------------------------------------------------------------
__global__ __launch_bounds__(256) void k_f32_to_bf16(const float* __restrict__ src,
                                                     __bf16* __restrict__ dst,
                                                     int n)
{
    for (int i = blockIdx.x * 256 + threadIdx.x; i < n; i += gridDim.x * 256)
        dst[i] = (__bf16)src[i];
}

// ---------------------------------------------------------------
// Kernel 3: bf16 WMMA GEMM  C[M,N](bf16) = A[M,K](bf16) @ W[K,N](bf16)
// 128x128 tile per block, BK=32, 8 waves; wave w owns rows [16w,16w+16)
// and all 128 cols (8 accumulator tiles).
// Double-buffered LDS pipeline: tile i+1 staged (A via TDM when available,
// B via cooperative transpose) while tile i feeds the WMMA pipe; TDM drained
// with s_wait_tensorcnt after the compute, single barrier per iteration.
// ---------------------------------------------------------------
#define BM 128
#define BN 128
#define BK 32

__device__ __forceinline__ void gemm_stage_A(const __bf16* __restrict__ A,
                                             __bf16* __restrict__ Asbuf,
                                             int k0, int rowBase, int M, int K, int tid)
{
#if HAVE_TDM
    if (tid < 32)
        tdm_load_2d(A + (size_t)rowBase * K + k0, Asbuf,
                    /*tile_cols*/ BK, /*tile_rows*/ BM,
                    /*tensor_cols*/ (unsigned)K, /*tensor_rows*/ (unsigned)M,
                    /*stride*/ (unsigned)K, /*elem*/ 1u);
#else
    const int i0 = tid, i1 = 256 + tid;
    uint4 a0 = *(const uint4*)(A + (size_t)(rowBase + (i0 >> 2)) * K + k0 + (i0 & 3) * 8);
    uint4 a1 = *(const uint4*)(A + (size_t)(rowBase + (i1 >> 2)) * K + k0 + (i1 & 3) * 8);
    *(uint4*)&Asbuf[(i0 >> 2) * BK + (i0 & 3) * 8] = a0;
    *(uint4*)&Asbuf[(i1 >> 2) * BK + (i1 & 3) * 8] = a1;
#endif
}

__device__ __forceinline__ void gemm_stage_B(const __bf16* __restrict__ W,
                                             __bf16* __restrict__ Bsbuf,
                                             int k0, int colBase, int N, int tid)
{
    const int i0 = tid, i1 = 256 + tid;
    uint4 b0 = *(const uint4*)(W + (size_t)(k0 + (i0 >> 4)) * N + colBase + (i0 & 15) * 8);
    uint4 b1 = *(const uint4*)(W + (size_t)(k0 + (i1 >> 4)) * N + colBase + (i1 & 15) * 8);
    const __bf16* e0 = (const __bf16*)&b0;
    const __bf16* e1 = (const __bf16*)&b1;
#pragma unroll
    for (int q = 0; q < 8; ++q) {
        Bsbuf[((i0 & 15) * 8 + q) * BK + (i0 >> 4)] = e0[q];
        Bsbuf[((i1 & 15) * 8 + q) * BK + (i1 >> 4)] = e1[q];
    }
}

__global__ __launch_bounds__(256) void k_wmma_gemm_bf16(const __bf16* __restrict__ A,
                                                        const __bf16* __restrict__ W,
                                                        __bf16* __restrict__ C,
                                                        int M, int N, int K)
{
    __shared__ __bf16 As[2][BM * BK];   // [row][32]   2 x 8 KB
    __shared__ __bf16 Bs[2][BN * BK];   // [col][32]   2 x 8 KB (transposed)

    const int tid  = threadIdx.x;
    const int lane = tid & 31;
    const int w    = tid >> 5;          // 0..7
    const int rowBase = blockIdx.y * BM;
    const int colBase = blockIdx.x * BN;

    v8f acc[8];
#pragma unroll
    for (int tn = 0; tn < 8; ++tn)
#pragma unroll
        for (int r = 0; r < 8; ++r) acc[tn][r] = 0.f;

    // ---- prologue: stage tile 0 into buffer 0 ----
    gemm_stage_A(A, As[0], 0, rowBase, M, K, tid);
    gemm_stage_B(W, Bs[0], 0, colBase, N, tid);
#if HAVE_TDM
    if (tid < 32) __builtin_amdgcn_s_wait_tensorcnt(0);
#endif
    __syncthreads();

    const int arow = w * 16 + (lane & 15);
    const int kb   = (lane >> 4) * 8;
    const int kbb  = (lane >> 4) * 8;

    for (int k0 = 0; k0 < K; k0 += BK) {
        const int  cur      = (k0 / BK) & 1;
        const int  nxt      = cur ^ 1;
        const bool has_next = (k0 + BK) < K;

        // ---- stage tile i+1 into the other buffer (overlaps with WMMA) ----
        if (has_next) {
            gemm_stage_A(A, As[nxt], k0 + BK, rowBase, M, K, tid);
            gemm_stage_B(W, Bs[nxt], k0 + BK, colBase, N, tid);
            if (k0 + 2 * BK < K) {
                __builtin_prefetch(A + (size_t)(rowBase + (tid >> 1)) * K + k0 + 2 * BK, 0, 1);
                __builtin_prefetch(W + (size_t)(k0 + 2 * BK + (tid & 31)) * N + colBase, 0, 1);
            }
        }

        // ---- compute from current buffer ----
        // A fragment: 16x32; lane<16: e0..7->K0..7, e8..15->K16..23; lanes>=16 +8
        const __bf16* Ac = As[cur];
        const __bf16* Bc = Bs[cur];
        V16Frag av;
        av.u[0] = *(const uint4*)&Ac[arow * BK + kb];
        av.u[1] = *(const uint4*)&Ac[arow * BK + kb + 16];
#pragma unroll
        for (int tn = 0; tn < 8; ++tn) {
            // B fragment: 32x16; lane<16: e0..7->K0..7, e8..15->K16..23; lanes>=16 +8
            const int col = tn * 16 + (lane & 15);
            V16Frag bv;
            bv.u[0] = *(const uint4*)&Bc[col * BK + kbb];
            bv.u[1] = *(const uint4*)&Bc[col * BK + kbb + 16];
            acc[tn] = __builtin_amdgcn_wmma_f32_16x16x32_bf16(
                false, av.v, false, bv.v, (short)0, acc[tn], false, false);
        }

        // ---- drain DMA for tile i+1, then publish it ----
#if HAVE_TDM
        if (has_next && tid < 32) __builtin_amdgcn_s_wait_tensorcnt(0);
#endif
        __syncthreads();
    }

    // ---- store: C/D layout lane<16 -> M=r, lane>=16 -> M=8+r ----
    const int mloc  = w * 16 + 8 * (lane >> 4);
    const int nlane = lane & 15;
#pragma unroll
    for (int tn = 0; tn < 8; ++tn)
#pragma unroll
        for (int r = 0; r < 8; ++r) {
            size_t idx = (size_t)(rowBase + mloc + r) * N + colBase + tn * 16 + nlane;
            C[idx] = (__bf16)acc[tn][r];
        }
}

// ---------------------------------------------------------------
// Kernel 4: g1[8192,16] = x(bf16) @ Wg1[2048,16] (f32 accumulate).
// ---------------------------------------------------------------
__global__ __launch_bounds__(256) void k_gate_rank16(const __bf16* __restrict__ x,
                                                     const float* __restrict__ Wg1,
                                                     float* __restrict__ g1)
{
    const int r    = threadIdx.x & 15;
    const int rowl = threadIdx.x >> 4;
    const int row  = blockIdx.x * 16 + rowl;
    const __bf16* xr = x + (size_t)row * D_;
    float acc = 0.f;
    for (int d = 0; d < D_; ++d)
        acc += (float)xr[d] * Wg1[d * R_ + r];
    g1[row * R_ + r] = acc;
}

// ---------------------------------------------------------------
// Kernel 5: gk[8192,1024] = log_sigmoid(g1 @ Wg2 + bg2) / 16  (f32)
// ---------------------------------------------------------------
__global__ __launch_bounds__(256) void k_gate_expand(const float* __restrict__ g1,
                                                     const float* __restrict__ Wg2,
                                                     const float* __restrict__ bg2,
                                                     float* __restrict__ gk)
{
    const int row = blockIdx.x;                 // 0..8191
    __shared__ float gr[R_];
    if (threadIdx.x < R_) gr[threadIdx.x] = g1[row * R_ + threadIdx.x];
    __syncthreads();
#pragma unroll
    for (int q = 0; q < 4; ++q) {
        int col = threadIdx.x + q * 256;
        float s = bg2[col];
#pragma unroll
        for (int r = 0; r < R_; ++r)
            s += gr[r] * Wg2[r * KDIM + col];
        float ls = fminf(s, 0.f) - log1pf(expf(-fabsf(s)));
        gk[(size_t)row * KDIM + col] = ls * (1.0f / GLN_);
    }
}

// ---------------------------------------------------------------
// Kernel 6: chunked decay scan.  One block per (b,h); 512 threads = 16 waves.
// Wave w owns S rows [16*(w/2), +16) x cols [128*(w%2), +128)  -> 8 WMMA tiles.
// S lives in VGPR accumulators across all 32 chunks.
// LDS: gcum 64x128 f32 | blast,cdec 128 f32 | At 128x64 bf16 | Bt 256x64 bf16
//      | Ks 64x128 bf16 (k chunk staging)
// ---------------------------------------------------------------
#define SM_GCUM   0
#define SM_BLAST  (CHUNK_ * DK_ * 4)                 // 32768
#define SM_CDEC   (SM_BLAST + DK_ * 4)               // 33280
#define SM_AT     (SM_CDEC + DK_ * 4)                // 33792
#define SM_BT     (SM_AT + DK_ * CHUNK_ * 2)         // 50176
#define SM_KS     (SM_BT + DV_ * CHUNK_ * 2)         // 82944
#define SM_TOTAL  (SM_KS + CHUNK_ * DK_ * 2)         // 99328

__global__ __launch_bounds__(512) void k_chunk_scan(const __bf16* __restrict__ kproj,
                                                    const __bf16* __restrict__ vproj,
                                                    const float* __restrict__ gk,
                                                    const float* __restrict__ S0,
                                                    float* __restrict__ out)
{
    extern __shared__ char smem[];
    float*  gcum  = (float*)(smem + SM_GCUM);        // [c][dk]
    float*  blast = (float*)(smem + SM_BLAST);       // [dk]
    float*  cdec  = (float*)(smem + SM_CDEC);        // [dk]
    __bf16* At    = (__bf16*)(smem + SM_AT);         // [dk][c]
    __bf16* Bt    = (__bf16*)(smem + SM_BT);         // [dv][c]
    __bf16* Ks    = (__bf16*)(smem + SM_KS);         // [c][dk]

    const int bh   = blockIdx.x;                     // 0..31
    const int b    = bh / H_;
    const int h    = bh % H_;
    const int tid  = threadIdx.x;
    const int lane = tid & 31;
    const int w    = tid >> 5;                       // 0..15
    const int m0   = (w >> 1) * 16;                  // row block
    const int cb   = (w & 1) * 128;                  // col block

    v8f acc[8];
    const int mlo = 8 * (lane >> 4);
    const int nl  = lane & 15;
#pragma unroll
    for (int tn = 0; tn < 8; ++tn)
#pragma unroll
        for (int r = 0; r < 8; ++r) {
            int M  = m0 + mlo + r;
            int Nc = cb + tn * 16 + nl;
            acc[tn][r] = S0[((size_t)bh * DK_ + M) * DV_ + Nc];
        }

    for (int nchunk = 0; nchunk < NC_; ++nchunk) {
        const size_t tbase = (size_t)b * T_ + (size_t)nchunk * CHUNK_;

#if HAVE_TDM
        // ---- gate chunk (64x128 f32) + k chunk (64x128 bf16) via TDM ----
        if (tid < 32) {
            tdm_load_2d(gk + (tbase)*KDIM + h * DK_, gcum,
                        DK_, CHUNK_, KDIM, BT_, KDIM, 2u);
            tdm_load_2d(kproj + (tbase)*KDIM + h * DK_, Ks,
                        DK_, CHUNK_, KDIM, BT_, KDIM, 1u);
        }
#else
        // ---- cooperative staging: gcum[c][dk] (f32) and Ks[c][dk] (bf16) ----
#pragma unroll
        for (int j = 0; j < 4; ++j) {
            int u   = j * 512 + tid;                 // 0..2047 (uint4 of f32)
            int c   = u >> 5;
            int dk0 = (u & 31) * 4;
            *(uint4*)&gcum[c * DK_ + dk0] =
                *(const uint4*)(gk + (tbase + c) * KDIM + h * DK_ + dk0);
        }
#pragma unroll
        for (int j = 0; j < 2; ++j) {
            int u   = j * 512 + tid;                 // 0..1023 (uint4 of bf16 x8)
            int c   = u >> 4;
            int dk0 = (u & 15) * 8;
            *(uint4*)&Ks[c * DK_ + dk0] =
                *(const uint4*)(kproj + (tbase + c) * KDIM + h * DK_ + dk0);
        }
#endif
        // ---- Bt[dv][c] = v chunk transposed (bf16 passthrough) ----
#pragma unroll
        for (int j = 0; j < 4; ++j) {
            int u   = j * 512 + tid;                 // 0..2047
            int c   = u >> 5;
            int dv0 = (u & 31) * 8;
            uint4 pk = *(const uint4*)(vproj + (tbase + c) * VDIM + h * DV_ + dv0);
            const __bf16* e = (const __bf16*)&pk;
#pragma unroll
            for (int q = 0; q < 8; ++q)
                Bt[(dv0 + q) * CHUNK_ + c] = e[q];
        }
        // prefetch next chunk's rows
        if (nchunk + 1 < NC_) {
            __builtin_prefetch(gk + (tbase + CHUNK_ + (tid >> 3)) * KDIM + h * DK_, 0, 1);
            __builtin_prefetch(vproj + (tbase + CHUNK_ + (tid >> 3)) * VDIM + h * DV_, 0, 1);
        }
#if HAVE_TDM
        if (tid < 32) __builtin_amdgcn_s_wait_tensorcnt(0);
#endif
        __syncthreads();

        // ---- inclusive cumsum along c, per dk column ----
        if (tid < DK_) {
            float s = 0.f;
            for (int c = 0; c < CHUNK_; ++c) {
                s += gcum[c * DK_ + tid];
                gcum[c * DK_ + tid] = s;
            }
            blast[tid] = s;
            cdec[tid]  = expf(s);                    // chunk_decay
        }
        __syncthreads();

        // ---- At[dk][c] = bf16( k * exp(blast - cumsum) ) ----
#pragma unroll
        for (int j = 0; j < 16; ++j) {
            int i  = j * 512 + tid;
            int c  = i >> 7;
            int dk = i & 127;
            float kv  = (float)Ks[c * DK_ + dk];
            float dec = expf(blast[dk] - gcum[c * DK_ + dk]);
            At[dk * CHUNK_ + c] = (__bf16)(kv * dec);
        }
        __syncthreads();

        // ---- S = S * chunk_decay (row-wise) ----
        float cd[8];
#pragma unroll
        for (int r = 0; r < 8; ++r) cd[r] = cdec[m0 + mlo + r];
#pragma unroll
        for (int tn = 0; tn < 8; ++tn)
#pragma unroll
            for (int r = 0; r < 8; ++r) acc[tn][r] *= cd[r];

        // ---- S += At(16x64 slice) @ Bt(64x16 tiles), K split 2x32 ----
#pragma unroll
        for (int ks = 0; ks < 2; ++ks) {
            const int kk   = ks * 32;
            const int arow = m0 + (lane & 15);
            const int kb   = kk + (lane >> 4) * 8;
            V16Frag av;
            av.u[0] = *(const uint4*)&At[arow * CHUNK_ + kb];
            av.u[1] = *(const uint4*)&At[arow * CHUNK_ + kb + 16];
#pragma unroll
            for (int tn = 0; tn < 8; ++tn) {
                const int dv  = cb + tn * 16 + (lane & 15);
                const int kbb = kk + (lane >> 4) * 8;
                V16Frag bv;
                bv.u[0] = *(const uint4*)&Bt[dv * CHUNK_ + kbb];
                bv.u[1] = *(const uint4*)&Bt[dv * CHUNK_ + kbb + 16];
                acc[tn] = __builtin_amdgcn_wmma_f32_16x16x32_bf16(
                    false, av.v, false, bv.v, (short)0, acc[tn], false, false);
            }
        }
        __syncthreads();   // LDS reused next chunk
    }

    // ---- store final state (f32) ----
#pragma unroll
    for (int tn = 0; tn < 8; ++tn)
#pragma unroll
        for (int r = 0; r < 8; ++r) {
            int M  = m0 + mlo + r;
            int Nc = cb + tn * 16 + nl;
            out[((size_t)bh * DK_ + M) * DV_ + Nc] = acc[tn][r];
        }
}

// ---------------------------------------------------------------
// Host launcher
// ---------------------------------------------------------------
extern "C" void kernel_launch(void* const* d_in, const int* in_sizes, int n_in,
                              void* d_out, int out_size, void* d_ws, size_t ws_size,
                              hipStream_t stream)
{
    (void)in_sizes; (void)n_in; (void)out_size; (void)ws_size;

    const float* seg   = (const float*)d_in[0];
    const float* nw    = (const float*)d_in[1];
    const float* Wk    = (const float*)d_in[2];
    const float* Wv    = (const float*)d_in[3];
    const float* Wg1   = (const float*)d_in[4];
    const float* Wg2   = (const float*)d_in[5];
    const float* bg2   = (const float*)d_in[6];
    const float* S0    = (const float*)d_in[7];
    float*       outp  = (float*)d_out;

    // workspace layout (bytes, 256-aligned)
    char* ws = (char*)d_ws;
    __bf16* x_bf   = (__bf16*)(ws);                                  // 8192*2048*2
    __bf16* wk_bf  = (__bf16*)(ws + 33554432);                       // 2048*1024*2
    __bf16* wv_bf  = (__bf16*)(ws + 37748736);                       // 2048*2048*2
    __bf16* kproj  = (__bf16*)(ws + 46137344);                       // 8192*1024*2
    __bf16* vproj  = (__bf16*)(ws + 62914560);                       // 8192*2048*2
    float*  g1     = (float*)(ws + 96468992);                        // 8192*16*4
    float*  gkf    = (float*)(ws + 96993280);                        // 8192*1024*4

    // allow >64KB dynamic LDS for the scan kernel (CDNA5 WGP has 320KB)
    (void)hipFuncSetAttribute((const void*)k_chunk_scan,
                              hipFuncAttributeMaxDynamicSharedMemorySize, SM_TOTAL);

    // 1. RMSNorm -> x (bf16)
    k_rmsnorm_bf16<<<BT_, 256, 0, stream>>>(seg, nw, x_bf);

    // 2. weight converts
    k_f32_to_bf16<<<2048, 256, 0, stream>>>(Wk, wk_bf, D_ * KDIM);
    k_f32_to_bf16<<<2048, 256, 0, stream>>>(Wv, wv_bf, D_ * VDIM);

    // 3. k = x @ Wk ; v = x @ Wv   (bf16 WMMA, f32 accumulate)
    {
        dim3 gk_grid(KDIM / BN, BT_ / BM);
        k_wmma_gemm_bf16<<<gk_grid, 256, 0, stream>>>(x_bf, wk_bf, kproj, BT_, KDIM, D_);
        dim3 gv_grid(VDIM / BN, BT_ / BM);
        k_wmma_gemm_bf16<<<gv_grid, 256, 0, stream>>>(x_bf, wv_bf, vproj, BT_, VDIM, D_);
    }

    // 4. g1 = x @ Wg1 (rank 16)
    k_gate_rank16<<<BT_ / 16, 256, 0, stream>>>(x_bf, Wg1, g1);

    // 5. gk = log_sigmoid(g1 @ Wg2 + bg2) / 16
    k_gate_expand<<<BT_, 256, 0, stream>>>(g1, Wg2, bg2, gkf);

    // 6. chunked decay scan -> final state
    k_chunk_scan<<<B_ * H_, 512, SM_TOTAL, stream>>>(kproj, vproj, gkf, S0, outp);
}
